// Attention_17643725652486
// MI455X (gfx1250) — matrix operand order repaired and checked by reading, not compiled, not run on hardware
//
#include <hip/hip_runtime.h>

// Attention: B=4, T=4096, D=768, DH=64. fp32 in/out, bf16 WMMA internally.

#define DMODEL 768
#define DHEAD  64
#define SEQ    4096
#define NROWS  16384   // B*T
#define NK0    24      // DMODEL/32

typedef __attribute__((ext_vector_type(16))) __bf16 v16bf;
typedef __attribute__((ext_vector_type(8)))  __bf16 v8bf;
typedef __attribute__((ext_vector_type(8)))  float  v8f;
typedef __attribute__((ext_vector_type(4)))  float  v4f;

static __device__ __forceinline__ __bf16 f2bf(float f) { return (__bf16)f; }

static __device__ __forceinline__ v16bf cat16(v8bf lo, v8bf hi) {
  return __builtin_shufflevector(lo, hi, 0, 1, 2, 3, 4, 5, 6, 7,
                                 8, 9, 10, 11, 12, 13, 14, 15);
}

static __device__ __forceinline__ v8f wmma_bf16(v16bf a, v16bf b, v8f c) {
  // D = A(16x32 bf16) * B(32x16 bf16) + C(16x16 f32)
  return __builtin_amdgcn_wmma_f32_16x16x32_bf16(false, a, false, b, (short)0, c,
                                                 false, false);
}

// Packed weight layout: wpack[mat][nt][k0idx][lane][16] bf16, element i =
// W[nt*16 + (lane&15)][k0idx*32 + (lane>>4)*16 + i]  (B-fragment order).
#define WP_KSTRIDE   (32 * 16)
#define WP_NTSTRIDE  (NK0 * WP_KSTRIDE)
#define WP_MATSTRIDE (4 * WP_NTSTRIDE)

// ---------------------------------------------------------------------------
// Kernel 0: one-shot weight repack f32 -> bf16 in B-fragment order.
// ---------------------------------------------------------------------------
__global__ __launch_bounds__(256)
void wpack_kernel(const float* __restrict__ Wq, const float* __restrict__ Wk,
                  const float* __restrict__ Wv, __bf16* __restrict__ wpack) {
  const int tid = blockIdx.x * 256 + threadIdx.x;
  if (tid >= 3 * 4 * NK0 * 32) return;
  const int lane  = tid & 31;
  const int t2    = tid >> 5;
  const int k0idx = t2 % NK0;
  const int t3    = t2 / NK0;
  const int nt    = t3 & 3;
  const int mat   = t3 >> 2;
  const float* W  = (mat == 0) ? Wq : (mat == 1) ? Wk : Wv;

  const int row  = nt * 16 + (lane & 15);
  const int col0 = k0idx * 32 + (lane >> 4) * 16;
  const float* wr = W + (size_t)row * DMODEL + col0;

  v16bf bf;
#pragma unroll
  for (int i = 0; i < 16; i += 4) {
    v4f w4 = *(const v4f*)(wr + i);
    bf[i]   = f2bf(w4[0]); bf[i+1] = f2bf(w4[1]);
    bf[i+2] = f2bf(w4[2]); bf[i+3] = f2bf(w4[3]);
  }
  *(v16bf*)(wpack + (size_t)tid * 16) = bf;
}

// ---------------------------------------------------------------------------
// Kernel 1: fused QKV projection using pre-packed bf16 weights.
//   Q = (x Wq^T + bq) * 0.125  -> qbuf [NROWS, 64] bf16 (scale folded in)
//   K =  x Wk^T + bk           -> kbuf [NROWS, 64] bf16
//   V =  x Wv^T + bv           -> vtbuf[B, 64, T]  bf16 (transposed!)
// All 12 B-fragments of a k-step are loaded into distinct registers before
// the 12 WMMAs so the loads can be clause-batched and overlapped.
// ---------------------------------------------------------------------------
__global__ __launch_bounds__(128)
void qkv_proj_kernel(const float* __restrict__ x,
                     const __bf16* __restrict__ wpack,
                     const float* __restrict__ bq, const float* __restrict__ bk,
                     const float* __restrict__ bv,
                     __bf16* __restrict__ qbuf, __bf16* __restrict__ kbuf,
                     __bf16* __restrict__ vtbuf) {
  const int lane  = threadIdx.x & 31;
  const int wave  = threadIdx.x >> 5;
  const int l16   = lane & 15;
  const int lhalf = lane >> 4;
  const int rowBase = blockIdx.x * 64 + wave * 16;

  v8f accQ[4] = {}, accK[4] = {}, accV[4] = {};

  const float* xrow = x + (size_t)(rowBase + l16) * DMODEL + lhalf * 16;
  const __bf16* wl  = wpack + (size_t)lane * 16;

  for (int k0idx = 0; k0idx < NK0; ++k0idx) {
    const int k0 = k0idx * 32;

    // Batch all loads of this k-step first (A from x, 12 B frags from wpack).
    v4f xa = *(const v4f*)(xrow + k0);
    v4f xb = *(const v4f*)(xrow + k0 + 4);
    v4f xc = *(const v4f*)(xrow + k0 + 8);
    v4f xd = *(const v4f*)(xrow + k0 + 12);

    const __bf16* wk0 = wl + (size_t)k0idx * WP_KSTRIDE;
    v16bf bw[12];
#pragma unroll
    for (int nt = 0; nt < 4; ++nt) {
      const __bf16* base = wk0 + (size_t)nt * WP_NTSTRIDE;
      bw[nt]     = *(const v16bf*)(base);
      bw[4 + nt] = *(const v16bf*)(base + WP_MATSTRIDE);
      bw[8 + nt] = *(const v16bf*)(base + 2 * WP_MATSTRIDE);
    }

    v16bf a;
#pragma unroll
    for (int i = 0; i < 4; ++i) {
      a[i]      = f2bf(xa[i]);  a[i + 4]  = f2bf(xb[i]);
      a[i + 8]  = f2bf(xc[i]);  a[i + 12] = f2bf(xd[i]);
    }

#pragma unroll
    for (int nt = 0; nt < 4; ++nt) {
      accQ[nt] = wmma_bf16(a, bw[nt],     accQ[nt]);
      accK[nt] = wmma_bf16(a, bw[4 + nt], accK[nt]);
      accV[nt] = wmma_bf16(a, bw[8 + nt], accV[nt]);
    }
  }

  // Epilogue: C layout -> lane holds (M = r + lhalf*8, N = l16) per VGPR r.
#pragma unroll
  for (int nt = 0; nt < 4; ++nt) {
    const int col = nt * 16 + l16;
    const float bqv = bq[col], bkv = bk[col], bvv = bv[col];
#pragma unroll
    for (int r = 0; r < 8; ++r) {
      const int row = rowBase + r + lhalf * 8;
      qbuf[(size_t)row * DHEAD + col] = f2bf((accQ[nt][r] + bqv) * 0.125f);
      kbuf[(size_t)row * DHEAD + col] = f2bf(accK[nt][r] + bkv);
      const int bb = row >> 12;
      const int t  = row & (SEQ - 1);
      vtbuf[((size_t)bb * DHEAD + col) * SEQ + t] = f2bf(accV[nt][r] + bvv);
    }
  }
}

// ---------------------------------------------------------------------------
// Kernel 2: flash attention, S^T / O^T formulation, split-K x2.
// Block = 256 threads = 8 waves: waves 0-3 -> keys [0,2048), waves 4-7 ->
// keys [2048,4096), same 4 query tiles. Partial (m, l, O^T) merged via LDS.
// Per wave: 16 queries, d_head=64, keys streamed in chunks of 32.
// ---------------------------------------------------------------------------
__global__ __launch_bounds__(256)
void flash_attn_kernel(const __bf16* __restrict__ qbuf,
                       const __bf16* __restrict__ kbuf,
                       const __bf16* __restrict__ vtbuf,
                       float* __restrict__ out) {
  __shared__ float smem[4][32][34];   // [qtile][lane][o(32), m, l]

  const int lane  = threadIdx.x & 31;
  const int wave  = threadIdx.x >> 5;
  const int l16   = lane & 15;
  const int lhalf = lane >> 4;
  const int qtile = wave & 3;
  const int half  = wave >> 2;                 // key-range half
  const int qBase = blockIdx.x * 64 + qtile * 16;
  const int b     = qBase >> 12;

  const __bf16* Qb = qbuf  + (size_t)qBase * DHEAD;
  const __bf16* Kb = kbuf  + (size_t)b * SEQ * DHEAD;
  const __bf16* Vt = vtbuf + (size_t)b * DHEAD * SEQ;

  // Q^T B-fragments (d x queries): lane -> col q=l16, d = c*32+lhalf*16+0..15.
  v16bf qf[2];
#pragma unroll
  for (int c = 0; c < 2; ++c)
    qf[c] = *(const v16bf*)(Qb + (size_t)l16 * DHEAD + c * 32 + lhalf * 16);

  v8f o[4] = {};             // O^T tiles: h = ht*16 + r + lhalf*8, q = l16
  float m = -3.0e38f, l = 0.0f;

  const int jBeg = half * (SEQ / 2);
  const int jEnd = jBeg + (SEQ / 2);

#pragma unroll 2
  for (int j0 = jBeg; j0 < jEnd; j0 += 32) {
    if (j0 + 32 < jEnd) {
      __builtin_prefetch(Kb + (size_t)(j0 + 32 + l16) * DHEAD, 0, 1);
      __builtin_prefetch(Vt + (size_t)l16 * SEQ + j0 + 32, 0, 1);
    }

    // ---- Issue ALL chunk loads up front: K frags AND V frags --------------
    v16bf kf[2][2];
#pragma unroll
    for (int t = 0; t < 2; ++t) {
#pragma unroll
      for (int c = 0; c < 2; ++c) {
        const __bf16* kr =
            Kb + (size_t)(j0 + t * 16 + l16) * DHEAD + c * 32 + lhalf * 16;
        kf[t][c] = cat16(*(const v8bf*)(kr), *(const v8bf*)(kr + 8));
      }
    }
    v16bf vf[4];
#pragma unroll
    for (int h = 0; h < 4; ++h) {
      const __bf16* vr = Vt + (size_t)(h * 16 + l16) * SEQ + j0 + lhalf * 16;
      vf[h] = cat16(*(const v8bf*)(vr), *(const v8bf*)(vr + 8));
    }

    // ---- S^T tiles (keys x queries); Q pre-scaled by 1/sqrt(64) -----------
    v8f s0 = {}, s1 = {};
    s0 = wmma_bf16(kf[0][0], qf[0], s0);
    s0 = wmma_bf16(kf[0][1], qf[1], s0);
    s1 = wmma_bf16(kf[1][0], qf[0], s1);
    s1 = wmma_bf16(kf[1][1], qf[1], s1);

    // ---- Online softmax: lane owns 16 of the chunk's 32 scores ------------
    float cmax = -3.0e38f;
#pragma unroll
    for (int r = 0; r < 8; ++r) cmax = fmaxf(cmax, fmaxf(s0[r], s1[r]));
    cmax = fmaxf(cmax, __shfl_xor(cmax, 16, 32));
    const float mnew = fmaxf(m, cmax);
    const float sc   = __expf(m - mnew);
    m = mnew;
    l *= sc;
    float p0[8], p1[8];
#pragma unroll
    for (int r = 0; r < 8; ++r) {
      p0[r] = __expf(s0[r] - mnew);
      p1[r] = __expf(s1[r] - mnew);
      l += p0[r] + p1[r];
    }
#pragma unroll
    for (int h = 0; h < 4; ++h)
#pragma unroll
      for (int r = 0; r < 8; ++r) o[h][r] *= sc;

    // ---- P^T B-fragment via one half-wave exchange ------------------------
    v16bf pb;
#pragma unroll
    for (int r = 0; r < 8; ++r) {
      const float send = lhalf ? p0[r] : p1[r];
      const float recv = __shfl_xor(send, 16, 32);
      const float e0 = lhalf ? recv  : p0[r];
      const float e1 = lhalf ? p1[r] : recv;
      pb[r]     = f2bf(e0);
      pb[r + 8] = f2bf(e1);
    }

    // ---- O^T += Vt-tile(16h x 32k) * P^T(32k x 16q) -----------------------
#pragma unroll
    for (int h = 0; h < 4; ++h) o[h] = wmma_bf16(vf[h], pb, o[h]);
  }

  // ---- Split-K merge via LDS ----------------------------------------------
  if (half == 1) {
#pragma unroll
    for (int h = 0; h < 4; ++h)
#pragma unroll
      for (int r = 0; r < 8; ++r) smem[qtile][lane][h * 8 + r] = o[h][r];
    smem[qtile][lane][32] = m;
    smem[qtile][lane][33] = l;
  }
  __syncthreads();
  if (half == 0) {
    const float m2 = smem[qtile][lane][32];
    const float l2 = smem[qtile][lane][33];
    const float mm = fmaxf(m, m2);
    const float a1 = __expf(m - mm);
    const float a2 = __expf(m2 - mm);
    l = l * a1 + l2 * a2;
#pragma unroll
    for (int h = 0; h < 4; ++h)
#pragma unroll
      for (int r = 0; r < 8; ++r)
        o[h][r] = o[h][r] * a1 + smem[qtile][lane][h * 8 + r] * a2;

    l += __shfl_xor(l, 16, 32);          // merge half-wave partner partials
    const float inv = 1.0f / l;
    float* orow = out + (size_t)(qBase + l16) * DHEAD;
#pragma unroll
    for (int h = 0; h < 4; ++h)
#pragma unroll
      for (int r = 0; r < 8; ++r)
        orow[h * 16 + lhalf * 8 + r] = o[h][r] * inv;
  }
}

// ---------------------------------------------------------------------------
extern "C" void kernel_launch(void* const* d_in, const int* in_sizes, int n_in,
                              void* d_out, int out_size, void* d_ws, size_t ws_size,
                              hipStream_t stream) {
  (void)in_sizes; (void)n_in; (void)out_size; (void)ws_size;
  const float* x  = (const float*)d_in[0];
  const float* Wq = (const float*)d_in[1];
  const float* bq = (const float*)d_in[2];
  const float* Wk = (const float*)d_in[3];
  const float* bk = (const float*)d_in[4];
  const float* Wv = (const float*)d_in[5];
  const float* bv = (const float*)d_in[6];

  __bf16* qbuf  = (__bf16*)d_ws;                         // 2 MB
  __bf16* kbuf  = qbuf + (size_t)NROWS * DHEAD;          // 2 MB
  __bf16* vtbuf = kbuf + (size_t)NROWS * DHEAD;          // 2 MB, [B,64,T]
  __bf16* wpack = vtbuf + (size_t)NROWS * DHEAD;         // 288 KB

  wpack_kernel<<<dim3(36), dim3(256), 0, stream>>>(Wq, Wk, Wv, wpack);
  qkv_proj_kernel<<<dim3(NROWS / 64), dim3(128), 0, stream>>>(
      x, wpack, bq, bk, bv, qbuf, kbuf, vtbuf);
  flash_attn_kernel<<<dim3(NROWS / 64), dim3(256), 0, stream>>>(
      qbuf, kbuf, vtbuf, (float*)d_out);
}
